// DeepDDS_Wang_71150428226164
// MI455X (gfx1250) — matrix operand order repaired and checked by reading, not compiled, not run on hardware
//
#include <hip/hip_runtime.h>
#include <hip/hip_bf16.h>

// ---------------------------------------------------------------------------
// Types for CDNA5 WMMA
// ---------------------------------------------------------------------------
typedef __attribute__((ext_vector_type(16))) __bf16 v16bf;
typedef __attribute__((ext_vector_type(8)))  __bf16 v8bf;
typedef __attribute__((ext_vector_type(8)))  float  v8f;

#define NPG   32      // nodes per graph
#define EPG   128     // edges per graph
#define NGR   8192    // graphs
#define NE    1048576 // edges total
#define FMAX  312     // max feature width in the GCN stack (row stride in LDS)
#define KPMAX 160     // max padded K inside the GCN stack
#define NPMAX 320     // max padded N inside the GCN stack

// ---------------------------------------------------------------------------
// Fused per-graph GCN branch: degree norm + 3 GCN layers + global max pool.
// One workgroup (256 thr / 8 waves) per graph; everything lives in LDS.
//   s_h  [32][FMAX] f32   current features -> becomes aggregation accumulator
//   s_xw [32][FMAX] f32   X @ W result, pre-aggregation
//   s_ha [32][Kpad] bf16  bf16 copy of features (A fragments, vector ds reads)
//   s_wt [Npad][32] bf16  transposed K-chunk of W (B fragments contiguous)
// ---------------------------------------------------------------------------
__global__ __launch_bounds__(256)
void fused_graph_kernel(const float* __restrict__ x, const int* __restrict__ eidx,
                        const float* __restrict__ W1, const float* __restrict__ b1,
                        const float* __restrict__ W2, const float* __restrict__ b2,
                        const float* __restrict__ W3, const float* __restrict__ b3,
                        float* __restrict__ pool)
{
    extern __shared__ char smem[];
    float*  s_h    = (float*)smem;                        // 32*312 f32   (39936 B)
    float*  s_xw   = s_h  + NPG * FMAX;                   // 32*312 f32   (39936 B)
    __bf16* s_ha   = (__bf16*)(s_xw + NPG * FMAX);        // 32*160 bf16  (10240 B)
    __bf16* s_wt   = s_ha + NPG * KPMAX;                  // 320*32 bf16  (20480 B)
    float*  s_dinv = (float*)(s_wt + NPMAX * 32);         // 32
    int*    s_row  = (int*)(s_dinv + NPG);                // 128
    int*    s_col  = s_row + EPG;                         // 128

    const int g    = blockIdx.x;
    const int tid  = threadIdx.x;
    const int lane = tid & 31;
    const int wave = tid >> 5;
    const int l    = lane & 15;
    const int hi   = lane >> 4;
    const int base = g * NPG;

    // ---- load 32x78 node features ----
    for (int i = tid; i < NPG * 78; i += 256) {
        int n = i / 78, f = i % 78;
        s_h[n * FMAX + f] = x[(base + n) * 78 + f];
    }
    // ---- load edges (local ids) ----
    if (tid < EPG) {
        s_row[tid] = eidx[g * EPG + tid] - base;          // edge_index[0] = row
        s_col[tid] = eidx[NE + g * EPG + tid] - base;     // edge_index[1] = col
    }
    // ---- degree: 1 (self loop) + #edges with row==n; dinv = rsqrt(deg) ----
    if (tid < NPG) s_dinv[tid] = 1.0f;
    __syncthreads();
    if (tid < EPG) atomicAdd(&s_dinv[s_row[tid]], 1.0f);
    __syncthreads();
    if (tid < NPG) s_dinv[tid] = rsqrtf(s_dinv[tid]);
    __syncthreads();

    const float* Ws[3] = {W1, W2, W3};
    const float* Bs[3] = {b1, b2, b3};
    const int    Ks[3] = {78, 78, 156};
    const int    Ns[3] = {78, 156, 312};

    for (int layer = 0; layer < 3; ++layer) {
        const float* Wm   = Ws[layer];
        const float* bias = Bs[layer];
        const int K      = Ks[layer];
        const int N      = Ns[layer];
        const int Kpad   = (K + 31) & ~31;                // 96 / 96 / 160
        const int ntiles = (N + 15) >> 4;
        const int Npad   = ntiles << 4;
        const int njobs  = 2 * ntiles;                    // 2 M-tiles (M=32) x ntiles

        // ---- bf16 copy of current features (zero K-pad) ----
        for (int n = 0; n < NPG; ++n)
            for (int k = tid; k < Kpad; k += 256)
                s_ha[n * Kpad + k] = (__bf16)((k < K) ? s_h[n * FMAX + k] : 0.0f);
        // (sync below, before first consume)

        v8f acc[5];                                       // up to 40 jobs / 8 waves
        #pragma unroll
        for (int s = 0; s < 5; ++s) { v8f z = {0,0,0,0,0,0,0,0}; acc[s] = z; }

        // ---- GEMM: s_xw = s_h @ W, K-chunked through LDS ----
        for (int kk = 0; kk < K; kk += 32) {
            __syncthreads();                              // s_wt reuse + s_ha ready
            // stage transposed W chunk: s_wt[n][kr] = W[kk+kr][n] (zero-padded)
            for (int kr = 0; kr < 32; ++kr) {
                for (int n = tid; n < Npad; n += 256) {
                    float v = (n < N && kk + kr < K) ? Wm[(kk + kr) * N + n] : 0.0f;
                    s_wt[n * 32 + kr] = (__bf16)v;
                }
            }
            if (kk + 32 < K)                              // prefetch next chunk
                __builtin_prefetch(&Wm[(kk + 32) * N], 0, 1);
            __syncthreads();
            #pragma unroll
            for (int s = 0; s < 5; ++s) {
                int j = wave + 8 * s;
                if (j < njobs) {                          // wave-uniform branch
                    int mtile = j & 1, ntile = j >> 1;
                    // A fragment: two 16B vector LDS loads
                    const __bf16* ap = s_ha + (mtile * 16 + l) * Kpad + kk + 16 * hi;
                    v8bf alo = *(const v8bf*)ap;
                    v8bf ahi = *(const v8bf*)(ap + 8);
                    v16bf a = __builtin_shufflevector(alo, ahi,
                              0,1,2,3,4,5,6,7,8,9,10,11,12,13,14,15);
                    // B fragment: one 32B vector LDS load (transposed layout)
                    v16bf b = *(const v16bf*)(s_wt + (ntile * 16 + l) * 32 + 16 * hi);
                    acc[s] = __builtin_amdgcn_wmma_f32_16x16x32_bf16(
                                 false, a, false, b, (short)0, acc[s], false, false);
                }
            }
        }
        // ---- write XW tiles to LDS ----
        #pragma unroll
        for (int s = 0; s < 5; ++s) {
            int j = wave + 8 * s;
            if (j < njobs) {
                int mtile = j & 1, ntile = j >> 1;
                int nn = ntile * 16 + l;
                if (nn < N) {
                    #pragma unroll
                    for (int r = 0; r < 8; ++r)
                        s_xw[(mtile * 16 + r + 8 * hi) * FMAX + nn] = acc[s][r];
                }
            }
        }
        __syncthreads();                                  // GEMM fully drained

        // ---- aggregation: self-loop init (s_h becomes accumulator) ----
        for (int n = 0; n < NPG; ++n)
            for (int f = tid; f < N; f += 256)
                s_h[n * FMAX + f] = s_xw[n * FMAX + f] * s_dinv[n] * s_dinv[n];
        __syncthreads();
        // ---- edge messages: column f only touched by thread (f mod 256) -> race free
        for (int e = 0; e < EPG; ++e) {
            int   r = s_row[e], c = s_col[e];
            float w = s_dinv[r] * s_dinv[c];
            for (int f = tid; f < N; f += 256)
                s_h[r * FMAX + f] += s_xw[c * FMAX + f] * w;
        }
        __syncthreads();
        // ---- + bias, ReLU ----
        for (int n = 0; n < NPG; ++n)
            for (int f = tid; f < N; f += 256) {
                float v = s_h[n * FMAX + f] + bias[f];
                s_h[n * FMAX + f] = v > 0.0f ? v : 0.0f;
            }
        __syncthreads();
    }

    // ---- global max pool over the 32 nodes ----
    for (int f = tid; f < FMAX; f += 256) {
        float m = s_h[f];
        for (int n = 1; n < NPG; ++n) {
            float v = s_h[n * FMAX + f];
            m = v > m ? v : m;
        }
        pool[g * FMAX + f] = m;
    }
}

// ---------------------------------------------------------------------------
// Weight prep: WT[Npad][Kpad] bf16 = transpose(W[K][N]), zero padded.
// ---------------------------------------------------------------------------
__global__ __launch_bounds__(256)
void prep_wt_kernel(const float* __restrict__ W, __bf16* __restrict__ WT,
                    int K, int N, int Kpad, int Npad)
{
    int i = blockIdx.x * 256 + threadIdx.x;
    if (i >= Npad * Kpad) return;
    int n = i / Kpad, k = i % Kpad;
    WT[i] = (__bf16)((n < N && k < K) ? W[k * N + n] : 0.0f);
}

// ---------------------------------------------------------------------------
// Generic WMMA GEMM:  C[M,N] = act(A[M,K] @ W[K,N] + bias)
// One wave per 16x16 C tile. B comes from pre-transposed bf16 WT[Npad][Kpad]:
// a B fragment is one contiguous 32B global load. A is fp32, vector-loaded
// (float4) on full chunks, converted with hardware bf16 casts.
// act: 0 = none, 1 = relu, 2 = prelu(*prelu_a)
// ---------------------------------------------------------------------------
__global__ __launch_bounds__(256)
void gemm_wmma(const float* __restrict__ A, const __bf16* __restrict__ WT,
               const float* __restrict__ bias, float* __restrict__ C,
               int M, int K, int Kpad, int N, int act,
               const float* __restrict__ prelu_a)
{
    const int lane  = threadIdx.x & 31;
    const int wave  = threadIdx.x >> 5;
    const int l     = lane & 15;
    const int hi    = lane >> 4;
    const int mtile = blockIdx.x;
    const int ntile = blockIdx.y * 8 + wave;
    const int ntiles = (N + 15) >> 4;
    if (ntile >= ntiles) return;                          // wave-uniform exit

    const int row = mtile * 16 + l;                       // M is a multiple of 16
    const int n   = ntile * 16 + l;
    const float* arow = A + (size_t)row * K;
    const __bf16* wrow = WT + (size_t)n * Kpad;

    v8f acc = {0,0,0,0,0,0,0,0};
    for (int kk = 0; kk < K; kk += 32) {
        v16bf a;
        if (kk + 32 <= K) {                               // full chunk: vector loads
            const float* ap = arow + kk + 16 * hi;
            float4 f0 = *(const float4*)(ap);
            float4 f1 = *(const float4*)(ap + 4);
            float4 f2 = *(const float4*)(ap + 8);
            float4 f3 = *(const float4*)(ap + 12);
            a[0]=(__bf16)f0.x; a[1]=(__bf16)f0.y; a[2]=(__bf16)f0.z; a[3]=(__bf16)f0.w;
            a[4]=(__bf16)f1.x; a[5]=(__bf16)f1.y; a[6]=(__bf16)f1.z; a[7]=(__bf16)f1.w;
            a[8]=(__bf16)f2.x; a[9]=(__bf16)f2.y; a[10]=(__bf16)f2.z; a[11]=(__bf16)f2.w;
            a[12]=(__bf16)f3.x; a[13]=(__bf16)f3.y; a[14]=(__bf16)f3.z; a[15]=(__bf16)f3.w;
        } else {                                          // K tail (e.g. K=1000)
            #pragma unroll
            for (int i = 0; i < 16; ++i) {
                int ka = kk + i + 16 * hi;
                a[i] = (__bf16)(ka < K ? arow[ka] : 0.0f);
            }
        }
        // B fragment: 16 contiguous bf16 (32B) from the transposed weights
        v16bf b = *(const v16bf*)(wrow + kk + 16 * hi);
        acc = __builtin_amdgcn_wmma_f32_16x16x32_bf16(
                  false, a, false, b, (short)0, acc, false, false);
    }

    if (n < N) {
        float pa = (act == 2) ? prelu_a[0] : 0.0f;
        float bv = bias[n];
        #pragma unroll
        for (int r = 0; r < 8; ++r) {
            int m = mtile * 16 + r + 8 * hi;
            float v = acc[r] + bv;
            if (act == 1)      v = v > 0.0f ? v : 0.0f;
            else if (act == 2) v = v >= 0.0f ? v : pa * v;
            C[(size_t)m * N + n] = v;
        }
    }
}

// ---------------------------------------------------------------------------
// Concat [g1|g2|c] (each 8192x128) and L2-normalize rows -> xc (8192x384)
// ---------------------------------------------------------------------------
__global__ __launch_bounds__(128)
void concat_norm_kernel(const float* __restrict__ g1, const float* __restrict__ g2,
                        const float* __restrict__ c, float* __restrict__ xc)
{
    __shared__ float red[128];
    const int row = blockIdx.x, t = threadIdx.x;
    float v0 = g1[row * 128 + t];
    float v1 = g2[row * 128 + t];
    float v2 = c [row * 128 + t];
    red[t] = v0 * v0 + v1 * v1 + v2 * v2;
    __syncthreads();
    for (int s = 64; s > 0; s >>= 1) {
        if (t < s) red[t] += red[t + s];
        __syncthreads();
    }
    float inv = 1.0f / fmaxf(sqrtf(red[0]), 1e-12f);
    xc[row * 384 +       t] = v0 * inv;
    xc[row * 384 + 128 + t] = v1 * inv;
    xc[row * 384 + 256 + t] = v2 * inv;
}

// ---------------------------------------------------------------------------
// Final: out = sigmoid(xc2[8192,128] @ Wo[128,1] + bo)
// ---------------------------------------------------------------------------
__global__ __launch_bounds__(256)
void final_kernel(const float* __restrict__ xc2, const float* __restrict__ Wo,
                  const float* __restrict__ bo, float* __restrict__ out)
{
    int row = blockIdx.x * blockDim.x + threadIdx.x;
    if (row >= NGR) return;
    float s = 0.0f;
    #pragma unroll 4
    for (int i = 0; i < 128; ++i) s += xc2[row * 128 + i] * Wo[i];
    s += bo[0];
    out[row] = 1.0f / (1.0f + __expf(-s));
}

// ---------------------------------------------------------------------------
// Launcher
// ---------------------------------------------------------------------------
extern "C" void kernel_launch(void* const* d_in, const int* in_sizes, int n_in,
                              void* d_out, int out_size, void* d_ws, size_t ws_size,
                              hipStream_t stream)
{
    const float* x1   = (const float*)d_in[0];
    const int*   e1   = (const int*)  d_in[1];
    const float* x2   = (const float*)d_in[2];
    const int*   e2   = (const int*)  d_in[3];
    const float* cell = (const float*)d_in[4];
    const float* W1 = (const float*)d_in[7],  *b1 = (const float*)d_in[8];
    const float* W2 = (const float*)d_in[9],  *b2 = (const float*)d_in[10];
    const float* W3 = (const float*)d_in[11], *b3 = (const float*)d_in[12];
    const float* Wg1= (const float*)d_in[13], *bg1= (const float*)d_in[14];
    const float* Wg2= (const float*)d_in[15], *bg2= (const float*)d_in[16];
    const float* Wr1= (const float*)d_in[17], *br1= (const float*)d_in[18];
    const float* Wr2= (const float*)d_in[19], *br2= (const float*)d_in[20];
    const float* Wr3= (const float*)d_in[21], *br3= (const float*)d_in[22];
    const float* Wf1= (const float*)d_in[23], *bf1= (const float*)d_in[24];
    const float* Wf2= (const float*)d_in[25], *bf2= (const float*)d_in[26];
    const float* Wo = (const float*)d_in[27], *bo = (const float*)d_in[28];
    const float* pa = (const float*)d_in[29];

    float* ws    = (float*)d_ws;
    float* pool1 = ws;                          // 8192*312
    float* pool2 = pool1 + (size_t)NGR * FMAX;  // 8192*312
    float* tA    = pool2 + (size_t)NGR * FMAX;  // 8192*512 (generic temp)
    float* tB    = tA    + (size_t)NGR * 512;   // 8192*512 (generic temp)
    float* g1o   = tB    + (size_t)NGR * 512;   // 8192*128
    float* g2o   = g1o   + (size_t)NGR * 128;
    float* co    = g2o   + (size_t)NGR * 128;
    float* xc    = co    + (size_t)NGR * 128;   // 8192*384
    float* xc2   = xc    + (size_t)NGR * 384;   // 8192*128

    // bf16 transposed weight staging (32B-aligned; all sizes are even floats)
    __bf16* wtp  = (__bf16*)(xc2 + (size_t)NGR * 128);
    auto carve = [&](int K, int N) {
        int Kpad = (K + 31) & ~31, Npad = (N + 15) & ~15;
        __bf16* p = wtp; wtp += (size_t)Kpad * Npad; return p;
    };
    __bf16* wg1T = carve(312, 156);   // Kpad 320, Npad 160
    __bf16* wg2T = carve(156, 128);   // 160 x 128
    __bf16* wr1T = carve(1000, 512);  // 1024 x 512
    __bf16* wr2T = carve(512, 256);
    __bf16* wr3T = carve(256, 128);
    __bf16* wf1T = carve(384, 512);
    __bf16* wf2T = carve(512, 128);

    const int ldsBytes = NPG * FMAX * 4 * 2     // s_h + s_xw (f32)
                       + NPG * KPMAX * 2        // s_ha (bf16)
                       + NPMAX * 32 * 2         // s_wt (bf16)
                       + NPG * 4                // s_dinv
                       + EPG * 4 * 2;           // s_row + s_col  (~112 KB)
    (void)hipFuncSetAttribute((const void*)fused_graph_kernel,
                              hipFuncAttributeMaxDynamicSharedMemorySize, ldsBytes);

    // ---- weight prep (cheap, L2-resident) ----
    auto prep = [&](const float* W, __bf16* WT, int K, int N) {
        int Kpad = (K + 31) & ~31, Npad = (N + 15) & ~15;
        int tot = Kpad * Npad;
        prep_wt_kernel<<<(tot + 255) / 256, 256, 0, stream>>>(W, WT, K, N, Kpad, Npad);
    };
    prep(Wg1, wg1T, 312, 156);
    prep(Wg2, wg2T, 156, 128);
    prep(Wr1, wr1T, 1000, 512);
    prep(Wr2, wr2T, 512, 256);
    prep(Wr3, wr3T, 256, 128);
    prep(Wf1, wf1T, 384, 512);
    prep(Wf2, wf2T, 512, 128);

    // ---- drug branches (shared weights, as in the reference forward) ----
    fused_graph_kernel<<<NGR, 256, ldsBytes, stream>>>(x1, e1, W1, b1, W2, b2, W3, b3, pool1);
    fused_graph_kernel<<<NGR, 256, ldsBytes, stream>>>(x2, e2, W1, b1, W2, b2, W3, b3, pool2);

    auto gemm = [&](const float* A, const __bf16* WT, const float* B, float* C,
                    int M, int K, int N, int act, const float* pr) {
        int Kpad = (K + 31) & ~31;
        dim3 grid(M / 16, (((N + 15) / 16) + 7) / 8);
        gemm_wmma<<<grid, 256, 0, stream>>>(A, WT, B, C, M, K, Kpad, N, act, pr);
    };

    // Graph-level FCs: 312 -> 156 (relu) -> 128
    gemm(pool1, wg1T, bg1, tA,  NGR, 312, 156, 1, nullptr);
    gemm(tA,    wg2T, bg2, g1o, NGR, 156, 128, 0, nullptr);
    gemm(pool2, wg1T, bg1, tA,  NGR, 312, 156, 1, nullptr);
    gemm(tA,    wg2T, bg2, g2o, NGR, 156, 128, 0, nullptr);

    // Cell branch: 1000 -> 512 (relu) -> 256 (relu) -> 128
    gemm(cell, wr1T, br1, tA, NGR, 1000, 512, 1, nullptr);
    gemm(tA,   wr2T, br2, tB, NGR, 512,  256, 1, nullptr);
    gemm(tB,   wr3T, br3, co, NGR, 256,  128, 0, nullptr);

    // Concat + L2 normalize
    concat_norm_kernel<<<NGR, 128, 0, stream>>>(g1o, g2o, co, xc);

    // Fusion MLP: 384 -> 512 (prelu) -> 128 (prelu) -> 1 (sigmoid)
    gemm(xc, wf1T, bf1, tA,  NGR, 384, 512, 2, pa);
    gemm(tA, wf2T, bf2, xc2, NGR, 512, 128, 2, pa);
    final_kernel<<<NGR / 256, 256, 0, stream>>>(xc2, Wo, bo, (float*)d_out);
}
